// Net_45767171506300
// MI455X (gfx1250) — matrix operand order repaired and checked by reading, not compiled, not run on hardware
//
#include <hip/hip_runtime.h>
#include <hip/hip_bf16.h>
#include <math.h>

#define N_NODES 20000
#define N_EDGES 640000
#define KSP 5
#define KK 25
#define CLASSES 6890

typedef __attribute__((ext_vector_type(2)))  float  v2f;
typedef __attribute__((ext_vector_type(8)))  float  v8f;
typedef __attribute__((ext_vector_type(16))) __bf16 v16bf;
typedef __attribute__((ext_vector_type(8)))  __bf16 v8bf;
typedef __attribute__((ext_vector_type(4)))  unsigned int v4u;
typedef __attribute__((ext_vector_type(8)))  int    v8i;
typedef __attribute__((ext_vector_type(4)))  int    v4i;

// ---------------------------------------------------------------------------
// Edge preprocessing: degree-1 B-spline basis (4 taps) and flat gather index
// gidx = src*KK + kidx, so H rows can be gathered directly.
// ---------------------------------------------------------------------------
__global__ void edge_prep(const int* __restrict__ ei, const float* __restrict__ pseudo,
                          float* __restrict__ basis, int* __restrict__ gidx) {
    int e = blockIdx.x * blockDim.x + threadIdx.x;
    if (e >= N_EDGES) return;
    int src = ei[e];
    float v0 = pseudo[2 * e + 0] * (float)(KSP - 1);
    float v1 = pseudo[2 * e + 1] * (float)(KSP - 1);
    float fi0 = fminf(fmaxf(floorf(v0), 0.f), (float)(KSP - 2));
    float fi1 = fminf(fmaxf(floorf(v1), 0.f), (float)(KSP - 2));
    float f0 = v0 - fi0, f1 = v1 - fi1;
    int i0 = (int)fi0, i1 = (int)fi1;
    int base = src * KK;
    float4 w = make_float4((1.f - f0) * (1.f - f1), f0 * (1.f - f1),
                           (1.f - f0) * f1,          f0 * f1);
    int4 k = make_int4(base + i0     + i1 * KSP,
                       base + i0 + 1 + i1 * KSP,
                       base + i0     + (i1 + 1) * KSP,
                       base + i0 + 1 + (i1 + 1) * KSP);
    *(float4*)(basis + 4 * e) = w;
    *(int4*)(gidx + 4 * e)    = k;
}

// ---------------------------------------------------------------------------
// Layer-1 H fill: cin == 1, H[n,k,c] = x[n] * W1[k,c]  (pure elementwise)
// ---------------------------------------------------------------------------
__global__ void h1_fill(const float* __restrict__ x, const float* __restrict__ W,
                        float* __restrict__ H, int cout) {
    long long t = (long long)blockIdx.x * blockDim.x + threadIdx.x;
    long long tot = (long long)N_NODES * KK * cout;
    if (t >= tot) return;
    int n = (int)(t / (KK * cout));
    int rem = (int)(t % (KK * cout));
    H[t] = x[n] * W[rem];
}

// ---------------------------------------------------------------------------
// H GEMM via f32 WMMA 16x16x4: for kernel tap k (blockIdx.y),
// H_k[16-node tile, 16-ch tile] = X[N,cin] @ W_k[cin,cout].  One wave / tile.
// ---------------------------------------------------------------------------
__global__ void h_gemm_f32(const float* __restrict__ X, const float* __restrict__ W,
                           float* __restrict__ H, int cin, int cout) {
    int lane = threadIdx.x;
    int g = lane >> 4, ln = lane & 15;
    int n0 = blockIdx.x * 16;
    int k  = blockIdx.y;
    int c0 = blockIdx.z * 16;
    const float* Wk = W + (size_t)k * cin * cout;
    int m = n0 + ln;
    v8f acc = {};
    for (int kk = 0; kk < cin; kk += 4) {
        v2f a, b;
        a.x = X[(size_t)m * cin + kk + 2 * g];
        a.y = X[(size_t)m * cin + kk + 2 * g + 1];
        b.x = Wk[(size_t)(kk + 2 * g)     * cout + c0 + ln];
        b.y = Wk[(size_t)(kk + 2 * g + 1) * cout + c0 + ln];
        acc = __builtin_amdgcn_wmma_f32_16x16x4_f32(false, a, false, b,
                                                    (short)0, acc, false, false);
    }
    for (int j = 0; j < 8; ++j) {
        int row = n0 + j + 8 * g;
        H[((size_t)row * KK + k) * cout + c0 + ln] = acc[j];
    }
}

// ---------------------------------------------------------------------------
// out[n,c] = bias[c] + sum_ci x[n,ci]*root[ci,c]   (seed for scatter-add)
// ---------------------------------------------------------------------------
__global__ void root_bias(const float* __restrict__ x, const float* __restrict__ root,
                          const float* __restrict__ b, float* __restrict__ out,
                          int cin, int cout) {
    long long t = (long long)blockIdx.x * blockDim.x + threadIdx.x;
    if (t >= (long long)N_NODES * cout) return;
    int n = (int)(t / cout), c = (int)(t % cout);
    float acc = b[c];
    for (int ci = 0; ci < cin; ++ci)
        acc += x[(size_t)n * cin + ci] * root[(size_t)ci * cout + c];
    out[t] = acc;
}

// ---------------------------------------------------------------------------
// Per-edge 4-tap basis-weighted gather of H rows, scatter-add into out[dst].
// ---------------------------------------------------------------------------
__global__ void edge_scatter(const float* __restrict__ basis, const int* __restrict__ gidx,
                             const int* __restrict__ dstp, const float* __restrict__ H,
                             float* __restrict__ out, int cout) {
    int groups = cout >> 2;
    long long t = (long long)blockIdx.x * blockDim.x + threadIdx.x;
    if (t >= (long long)N_EDGES * groups) return;
    int e  = (int)(t / groups);
    int cg = (int)(t % groups) * 4;
    float4 w = *(const float4*)(basis + 4 * e);
    int4   k = *(const int4*)(gidx + 4 * e);
    float4 h0 = *(const float4*)(H + (size_t)k.x * cout + cg);
    float4 h1 = *(const float4*)(H + (size_t)k.y * cout + cg);
    float4 h2 = *(const float4*)(H + (size_t)k.z * cout + cg);
    float4 h3 = *(const float4*)(H + (size_t)k.w * cout + cg);
    float m0 = w.x * h0.x + w.y * h1.x + w.z * h2.x + w.w * h3.x;
    float m1 = w.x * h0.y + w.y * h1.y + w.z * h2.y + w.w * h3.y;
    float m2 = w.x * h0.z + w.y * h1.z + w.z * h2.z + w.w * h3.z;
    float m3 = w.x * h0.w + w.y * h1.w + w.z * h2.w + w.w * h3.w;
    float* o = out + (size_t)dstp[e] * cout + cg;
    atomicAdd(o + 0, m0);
    atomicAdd(o + 1, m1);
    atomicAdd(o + 2, m2);
    atomicAdd(o + 3, m3);
}

__global__ void elu_k(float* __restrict__ v, long long n) {
    long long t = (long long)blockIdx.x * blockDim.x + threadIdx.x;
    if (t >= n) return;
    float x = v[t];
    v[t] = x > 0.f ? x : (expf(x) - 1.f);
}

__global__ void to_bf16(const float* __restrict__ in, __bf16* __restrict__ out, long long n) {
    long long t = (long long)blockIdx.x * blockDim.x + threadIdx.x;
    if (t >= n) return;
    out[t] = (__bf16)in[t];
}

// ---------------------------------------------------------------------------
// lin1: h4[N,64]bf16 @ W[64,256]bf16 + b -> ELU -> bf16   (one wave / tile)
// ---------------------------------------------------------------------------
__global__ void lin1_gemm(const __bf16* __restrict__ A, const __bf16* __restrict__ B,
                          const float* __restrict__ bias, __bf16* __restrict__ out) {
    const int Kd = 64, C = 256;
    int lane = threadIdx.x, g = lane >> 4, ln = lane & 15;
    int n0 = blockIdx.x * 16, c0 = blockIdx.y * 16;
    v8f acc = {};
    for (int kk = 0; kk < Kd; kk += 32) {
        v16bf a, b;
        const __bf16* ar = A + (size_t)(n0 + ln) * Kd + kk + (g ? 16 : 0);
        *(v8bf*)&a         = *(const v8bf*)ar;
        *(((v8bf*)&a) + 1) = *(const v8bf*)(ar + 8);
        for (int v = 0; v < 16; ++v)
            b[v] = B[(size_t)(kk + v + 16 * g) * C + c0 + ln];
        acc = __builtin_amdgcn_wmma_f32_16x16x32_bf16(false, a, false, b,
                                                      (short)0, acc, false, false);
    }
    for (int j = 0; j < 8; ++j) {
        int row = n0 + j + 8 * g;
        float v = acc[j] + bias[c0 + ln];
        v = v > 0.f ? v : (expf(v) - 1.f);
        out[(size_t)row * C + c0 + ln] = (__bf16)v;
    }
}

// ---------------------------------------------------------------------------
// lin2: hmid[N,256]bf16 @ W[256,6890]bf16 + b -> f32 logits.
// Block = 4 waves, tile = 64 rows x 32 cols, full K=256 panel in LDS.
//  - A panel staged by the Tensor Data Mover (TENSORcnt) with pad_enable:
//    16B pad after each 512B row => LDS row stride 528B (4-bank skew) =>
//    conflict-free ds_load_b128 operand reads.  tensor_dim1=N gives zero-fill
//    for the ragged last row block.
//  - B panel staged cooperatively, stored K-contiguous per column, so the
//    WMMA B operand (K = kk+v+16g at fixed col) is two contiguous b128 reads.
// ---------------------------------------------------------------------------
#define L2_ROWS 64
#define L2_COLS 32
#define L2_KD   256
#define PSTRIDE 264   // bf16 elems per LDS row: 528B = 16B-aligned, 4-bank skew

__global__ void __launch_bounds__(128)
lin2_gemm(const __bf16* __restrict__ A, const __bf16* __restrict__ B,
          const float* __restrict__ bias, float* __restrict__ out) {
    const int C = CLASSES;
    __shared__ __bf16 As[L2_ROWS * PSTRIDE];   // 33 KB, padded by TDM
    __shared__ __bf16 Bs[L2_COLS * PSTRIDE];   // 16.5 KB, [col][k] transposed

    int t    = threadIdx.x;
    int wv   = t >> 5;
    int lane = t & 31;
    int g = lane >> 4, ln = lane & 15;
    int n0 = blockIdx.x * L2_ROWS;
    int c0 = blockIdx.y * L2_COLS;

#if __has_builtin(__builtin_amdgcn_tensor_load_to_lds)
    // --- A panel via TDM: 2-D tile 256 x 64 rows, data_size=2B, padded ---
    if (t < 32) {
        unsigned lds_base = (unsigned)(unsigned long long)(uintptr_t)&As[0];
        unsigned long long ga =
            (unsigned long long)(uintptr_t)(A + (size_t)n0 * L2_KD);
        v4u g0;
        g0.x = 1u;                                   // count=1, user D#
        g0.y = lds_base;                             // lds_addr
        g0.z = (unsigned)ga;                         // global_addr[31:0]
        g0.w = (unsigned)((ga >> 32) & 0x01FFFFFFu)  // global_addr[56:32]
             | (2u << 30);                           // type=2 (image)
        v8i g1;
        g1[0] = (int)((1u << 16)      // data_size = 2 bytes
                    | (1u << 20)      // pad_enable
                    | (6u << 22)      // pad_interval: 128 DWORDs = 512B
                    | (3u << 25));    // pad_amount:   4 DWORDs  = 16B
        g1[1] = (int)(((unsigned)L2_KD & 0xFFFFu) << 16);        // tensor_dim0 lo
        g1[2] = (int)(((unsigned)N_NODES & 0xFFFFu) << 16);      // dim0 hi | dim1 lo
        g1[3] = (int)(((unsigned)L2_KD) << 16);                  // dim1 hi | tile_dim0
        g1[4] = (int)L2_ROWS;                                    // tile_dim1 | tile_dim2=0
        g1[5] = (int)L2_KD;                                      // tensor_dim0_stride lo
        g1[6] = 0;
        g1[7] = 0;
        v4i gz4 = {0, 0, 0, 0};
        v8i gz8 = {0, 0, 0, 0, 0, 0, 0, 0};
        __builtin_amdgcn_tensor_load_to_lds(g0, g1, gz4, gz4, gz8, 0);
    }
#else
    // --- fallback: cooperative padded fill of the A panel ---
    for (int idx = t; idx < L2_ROWS * L2_KD; idx += 128) {
        int r = idx >> 8, k = idx & (L2_KD - 1);
        int row = n0 + r;
        As[r * PSTRIDE + k] = (row < N_NODES) ? A[(size_t)row * L2_KD + k]
                                              : (__bf16)0.0f;
    }
#endif

    // --- B panel: coalesced global reads, transposed [col][k] LDS writes ---
    {
        int col = t & (L2_COLS - 1);
        int gc  = c0 + col;
        bool cok = gc < C;
        for (int k = t >> 5; k < L2_KD; k += 4)
            Bs[col * PSTRIDE + k] = cok ? B[(size_t)k * C + gc] : (__bf16)0.0f;
    }

#if __has_builtin(__builtin_amdgcn_tensor_load_to_lds)
#if __has_builtin(__builtin_amdgcn_s_wait_tensorcnt)
    if (t < 32) __builtin_amdgcn_s_wait_tensorcnt(0);
#endif
#endif
    __syncthreads();

    v8f acc0 = {}, acc1 = {};
    for (int kk = 0; kk < L2_KD; kk += 32) {
        v16bf a, b0, b1;
        const __bf16* ar = &As[(wv * 16 + ln) * PSTRIDE + kk + 16 * g];
        *(v8bf*)&a         = *(const v8bf*)ar;
        *(((v8bf*)&a) + 1) = *(const v8bf*)(ar + 8);
        const __bf16* br0 = &Bs[ln * PSTRIDE + kk + 16 * g];
        const __bf16* br1 = &Bs[(16 + ln) * PSTRIDE + kk + 16 * g];
        *(v8bf*)&b0         = *(const v8bf*)br0;
        *(((v8bf*)&b0) + 1) = *(const v8bf*)(br0 + 8);
        *(v8bf*)&b1         = *(const v8bf*)br1;
        *(((v8bf*)&b1) + 1) = *(const v8bf*)(br1 + 8);
        acc0 = __builtin_amdgcn_wmma_f32_16x16x32_bf16(false, a, false, b0,
                                                       (short)0, acc0, false, false);
        acc1 = __builtin_amdgcn_wmma_f32_16x16x32_bf16(false, a, false, b1,
                                                       (short)0, acc1, false, false);
    }

    int rowbase = n0 + wv * 16;
    for (int ct = 0; ct < 2; ++ct) {
        int col = c0 + ct * 16 + ln;
        if (col >= C) continue;
        float bb = bias[col];
        const v8f& acc = ct ? acc1 : acc0;
        for (int j = 0; j < 8; ++j) {
            int row = rowbase + j + 8 * g;
            if (row < N_NODES)
                out[(size_t)row * C + col] = acc[j] + bb;
        }
    }
}

// ---------------------------------------------------------------------------
// In-place row-wise log_softmax over 6890 classes. One block per row.
// ---------------------------------------------------------------------------
__global__ void log_softmax_k(float* __restrict__ out) {
    int row = blockIdx.x;
    float* r = out + (size_t)row * CLASSES;
    __shared__ float red[256];
    int t = threadIdx.x;
    float mx = -INFINITY;
    for (int c = t; c < CLASSES; c += 256) mx = fmaxf(mx, r[c]);
    red[t] = mx; __syncthreads();
    for (int s = 128; s > 0; s >>= 1) {
        if (t < s) red[t] = fmaxf(red[t], red[t + s]);
        __syncthreads();
    }
    mx = red[0]; __syncthreads();
    float sum = 0.f;
    for (int c = t; c < CLASSES; c += 256) sum += expf(r[c] - mx);
    red[t] = sum; __syncthreads();
    for (int s = 128; s > 0; s >>= 1) {
        if (t < s) red[t] += red[t + s];
        __syncthreads();
    }
    float lse = logf(red[0]) + mx;
    for (int c = t; c < CLASSES; c += 256) r[c] = r[c] - lse;
}

// ---------------------------------------------------------------------------
extern "C" void kernel_launch(void* const* d_in, const int* in_sizes, int n_in,
                              void* d_out, int out_size, void* d_ws, size_t ws_size,
                              hipStream_t stream) {
    const float* x      = (const float*)d_in[0];
    const int*   ei     = (const int*)  d_in[1];
    const float* pseudo = (const float*)d_in[2];
    const float* W1 = (const float*)d_in[3];  const float* root1 = (const float*)d_in[4];  const float* b1 = (const float*)d_in[5];
    const float* W2 = (const float*)d_in[6];  const float* root2 = (const float*)d_in[7];  const float* b2 = (const float*)d_in[8];
    const float* W3 = (const float*)d_in[9];  const float* root3 = (const float*)d_in[10]; const float* b3 = (const float*)d_in[11];
    const float* W4 = (const float*)d_in[12]; const float* root4 = (const float*)d_in[13]; const float* b4 = (const float*)d_in[14];
    const float* lin1_w = (const float*)d_in[15]; const float* lin1_b = (const float*)d_in[16];
    const float* lin2_w = (const float*)d_in[17]; const float* lin2_b = (const float*)d_in[18];

    // bump-allocate workspace (256B aligned)
    char* wp = (char*)d_ws;
    auto alloc = [&](size_t bytes) -> void* {
        void* p = (void*)wp;
        wp += (bytes + 255) & ~(size_t)255;
        return p;
    };
    float*  basis  = (float*) alloc((size_t)N_EDGES * 4 * sizeof(float));
    int*    gidx   = (int*)   alloc((size_t)N_EDGES * 4 * sizeof(int));
    float*  H      = (float*) alloc((size_t)N_NODES * KK * 64 * sizeof(float));
    float*  bufA   = (float*) alloc((size_t)N_NODES * 64 * sizeof(float));
    float*  bufB   = (float*) alloc((size_t)N_NODES * 64 * sizeof(float));
    __bf16* h4bf   = (__bf16*)alloc((size_t)N_NODES * 64 * sizeof(__bf16));
    __bf16* w1bf   = (__bf16*)alloc((size_t)64 * 256 * sizeof(__bf16));
    __bf16* hmidbf = (__bf16*)alloc((size_t)N_NODES * 256 * sizeof(__bf16));
    __bf16* w2bf   = (__bf16*)alloc((size_t)256 * CLASSES * sizeof(__bf16));
    (void)ws_size; (void)in_sizes; (void)n_in; (void)out_size;

    const int* dstp = ei + N_EDGES;

    edge_prep<<<(N_EDGES + 255) / 256, 256, 0, stream>>>(ei, pseudo, basis, gidx);

    auto conv_layer = [&](const float* xin, int cin, const float* Wl,
                          const float* rootl, const float* bl,
                          float* xout, int cout, bool layer1) {
        if (layer1) {
            long long tot = (long long)N_NODES * KK * cout;
            h1_fill<<<(unsigned)((tot + 255) / 256), 256, 0, stream>>>(xin, Wl, H, cout);
        } else {
            dim3 grid(N_NODES / 16, KK, cout / 16);
            h_gemm_f32<<<grid, 32, 0, stream>>>(xin, Wl, H, cin, cout);
        }
        long long nt = (long long)N_NODES * cout;
        root_bias<<<(unsigned)((nt + 255) / 256), 256, 0, stream>>>(xin, rootl, bl, xout, cin, cout);
        long long st = (long long)N_EDGES * (cout / 4);
        edge_scatter<<<(unsigned)((st + 255) / 256), 256, 0, stream>>>(basis, gidx, dstp, H, xout, cout);
        elu_k<<<(unsigned)((nt + 255) / 256), 256, 0, stream>>>(xout, nt);
    };

    conv_layer(x,    1,  W1, root1, b1, bufB, 32, true);
    conv_layer(bufB, 32, W2, root2, b2, bufA, 64, false);
    conv_layer(bufA, 64, W3, root3, b3, bufB, 64, false);
    conv_layer(bufB, 64, W4, root4, b4, bufA, 64, false);

    long long n64 = (long long)N_NODES * 64;
    to_bf16<<<(unsigned)((n64 + 255) / 256), 256, 0, stream>>>(bufA, h4bf, n64);
    to_bf16<<<(64 * 256 + 255) / 256, 256, 0, stream>>>(lin1_w, w1bf, 64 * 256);
    long long w2n = (long long)256 * CLASSES;
    to_bf16<<<(unsigned)((w2n + 255) / 256), 256, 0, stream>>>(lin2_w, w2bf, w2n);

    lin1_gemm<<<dim3(N_NODES / 16, 256 / 16), 32, 0, stream>>>(h4bf, w1bf, lin1_b, hmidbf);
    lin2_gemm<<<dim3((N_NODES + L2_ROWS - 1) / L2_ROWS,
                     (CLASSES + L2_COLS - 1) / L2_COLS), 128, 0, stream>>>(
        hmidbf, w2bf, lin2_b, (float*)d_out);
    log_softmax_k<<<N_NODES, 256, 0, stream>>>((float*)d_out);
}
